// Model_39676907885168
// MI455X (gfx1250) — matrix operand order repaired and checked by reading, not compiled, not run on hardware
//
#include <hip/hip_runtime.h>
#include <hip/hip_bf16.h>

#define NN 384
#define KT 12      // 384 / 32 k-steps per WMMA sweep
#define NT 24      // 384 / 16 n-tiles

typedef __attribute__((ext_vector_type(16))) __bf16        v16bf;
typedef __attribute__((ext_vector_type(2)))  __bf16        v2bf;
typedef __attribute__((ext_vector_type(8)))  float         v8f;
typedef __attribute__((ext_vector_type(8)))  unsigned int  v8u;

// Hardware float->bf16 conversion (v_cvt_pk_bf16_f32 on gfx1250).
__device__ __forceinline__ unsigned int pk(float lo, float hi) {
    v2bf t;
    t[0] = (__bf16)lo;
    t[1] = (__bf16)hi;
    return __builtin_bit_cast(unsigned int, t);
}
__device__ __forceinline__ unsigned short bf1(float x) {
    return __builtin_bit_cast(unsigned short, (__bf16)x);
}
__device__ __forceinline__ v16bf asbf(v8u x) { return __builtin_bit_cast(v16bf, x); }

// ---- fragment builders (k order used by this code for wave32 WMMA 16x16x32 bf16; A and B match) ----------
// A (16x32, 16-bit): lane row = lane&15; VGPR j holds K pair 16*(lane>>4) + 2j
// B (32x16, 16-bit): lane col = lane&15; VGPR j holds K pair 16*(lane>>4) + 2j
// C (16x16, f32)   : VGPR r -> M = r + 8*(lane>>4), N = lane&15

__device__ __forceinline__ v16bf load_a_f32(const float* __restrict__ rowp, int k0, int half) {
    v8u t;
#pragma unroll
    for (int j = 0; j < 8; ++j) {
        int kb = k0 + half * 16 + 2 * j;
        float2 f = *(const float2*)(rowp + kb);
        t[j] = pk(f.x, f.y);
    }
    return asbf(t);
}

__device__ __forceinline__ v16bf load_b_f32rows(const float* __restrict__ rowp, int k0, int half) {
    v8u t;
    int ks = k0 + half * 16;
#pragma unroll
    for (int j = 0; j < 8; ++j) {
        float2 f = *(const float2*)(rowp + ks + 2 * j);
        t[j] = pk(f.x, f.y);
    }
    return asbf(t);
}

// Contiguous bf16 row -> B fragment: pure 2x b128 loads, no VALU.
__device__ __forceinline__ v16bf load_b_bf16rows(const unsigned short* __restrict__ rowp, int k0, int half) {
    const unsigned int* p = (const unsigned int*)(rowp + k0 + half * 16);
    v8u t;
#pragma unroll
    for (int j = 0; j < 8; ++j) t[j] = p[j];
    return asbf(t);
}

__device__ __forceinline__ v16bf load_a_bf16row(const unsigned short* rowp, int k0, int half) {
    v8u t;
#pragma unroll
    for (int j = 0; j < 8; ++j) {
        int kb = k0 + half * 16 + 2 * j;
        t[j] = *(const unsigned int*)(rowp + kb);
    }
    return asbf(t);
}

// Fallback: strided f32 columns + cvt_pk (used only if workspace is too small).
__device__ __forceinline__ v16bf load_b_f32cols(const float* __restrict__ colp, int k0, int half) {
    v8u t;
    int ks = k0 + half * 16;
#pragma unroll
    for (int j = 0; j < 8; ++j) {
        float lo = colp[(size_t)(ks + 2 * j) * NN];
        float hi = colp[(size_t)(ks + 2 * j + 1) * NN];
        t[j] = pk(lo, hi);
    }
    return asbf(t);
}

// ---------------- kernel 1: v1 = x1 @ W^T + b  -> bf16 workspace ------------
__global__ __launch_bounds__(32) void linear_wmma(const float* __restrict__ x1,
                                                  const float* __restrict__ W,
                                                  const float* __restrict__ bias,
                                                  unsigned short* __restrict__ v1w) {
    const int lane = threadIdx.x;
    const int half = lane >> 4, l16 = lane & 15;
    const int i0 = blockIdx.x * 16, n0 = blockIdx.y * 16;
    const float* arow = x1 + (size_t)(i0 + l16) * NN;
    const float* brow = W  + (size_t)(n0 + l16) * NN;

    v8f acc = {};
#pragma unroll
    for (int kt = 0; kt < KT; ++kt) {
        v16bf a = load_a_f32(arow, kt * 32, half);
        v16bf b = load_b_f32rows(brow, kt * 32, half);
        acc = __builtin_amdgcn_wmma_f32_16x16x32_bf16(false, a, false, b,
                                                      (short)0, acc, false, false);
    }
    float bj = bias[n0 + l16];
#pragma unroll
    for (int r = 0; r < 8; ++r)
        v1w[(size_t)(i0 + r + half * 8) * NN + n0 + l16] = bf1(acc[r] + bj);
}

// -------- kernel 2: per-batch transpose + f32->bf16:  x2t[b][n][k] ----------
// Loads coalesced b32, stores coalesced b32 (two packed bf16 k-elements).
__global__ __launch_bounds__(256) void transpose_bf16(const float* __restrict__ x2,
                                                      unsigned short* __restrict__ x2t) {
    __shared__ float tile[32][33];
    const int b  = blockIdx.z;
    const int k0 = blockIdx.x * 32;
    const int n0 = blockIdx.y * 32;

    // load phase: 32x32 f32 tile, rows k, cols n
    {
        const int tx = threadIdx.x & 31;
        const int ty = threadIdx.x >> 5;          // 0..7
        const float* src = x2 + ((size_t)b * NN + k0) * NN + n0;
#pragma unroll
        for (int i = 0; i < 32; i += 8)
            tile[ty + i][tx] = src[(size_t)(ty + i) * NN + tx];
    }
    __syncthreads();

    // store phase: rows n, cols k; each thread packs two consecutive k into b32
    {
        const int sx = threadIdx.x & 15;          // k-pair index 0..15
        const int sy = threadIdx.x >> 4;          // 0..15 (row n within tile)
        unsigned int* dst = (unsigned int*)(x2t + ((size_t)b * NN + n0) * NN + k0);
#pragma unroll
        for (int i = 0; i < 32; i += 16) {
            int rr = sy + i;
            dst[(size_t)rr * (NN / 2) + sx] = pk(tile[2 * sx][rr], tile[2 * sx + 1][rr]);
        }
    }
}

// ------- kernel 3: per (batch, 16-row tile): S = Q@v1^T, softmax, O = P@V ---
template <bool TRANS>
__global__ __launch_bounds__(32) void fused_attn(const float* __restrict__ x2,
                                                 const unsigned short* __restrict__ x2t,
                                                 const unsigned short* __restrict__ v1w,
                                                 float* __restrict__ out) {
    __shared__ float          Ssm[16 * NN];   // 24 KB score strip (f32)
    __shared__ unsigned short Psm[16 * NN];   // 12 KB prob strip (bf16)

    const int lane = threadIdx.x;
    const int half = lane >> 4, l16 = lane & 15;
    const int b  = blockIdx.y;
    const int i0 = blockIdx.x * 16;

    const float* x2b  = x2 + (size_t)b * NN * NN;
    const float* qrow = x2b + (size_t)(i0 + l16) * NN;

    // Preload all 12 Q A-fragments (f32 -> bf16 once per wave).
    v16bf qa[KT];
#pragma unroll
    for (int kt = 0; kt < KT; ++kt) qa[kt] = load_a_f32(qrow, kt * 32, half);

    // ---- pass 1: scores S[i, :] = Q @ v1^T into LDS ----
    for (int nt = 0; nt < NT; ++nt) {
        const unsigned short* vrow = v1w + (size_t)(nt * 16 + l16) * NN;
        v8f acc = {};
#pragma unroll
        for (int kt = 0; kt < KT; ++kt) {
            v16bf bfrag = load_b_bf16rows(vrow, kt * 32, half);
            acc = __builtin_amdgcn_wmma_f32_16x16x32_bf16(false, qa[kt], false, bfrag,
                                                          (short)0, acc, false, false);
        }
#pragma unroll
        for (int r = 0; r < 8; ++r)
            Ssm[(r + half * 8) * NN + nt * 16 + l16] = acc[r];
    }

    // ---- row softmax of S/1000 (2 lanes per row, combine via shfl_xor 16) ----
    {
        float* srow = &Ssm[l16 * NN];
        const float inv = 1.0f / 1000.0f;
        float mx = -3.0e38f;
        for (int c = half; c < NN; c += 2) mx = fmaxf(mx, srow[c]);
        mx = fmaxf(mx, __shfl_xor(mx, 16, 32));
        float sum = 0.0f;
        for (int c = half; c < NN; c += 2) {
            float e = __expf((srow[c] - mx) * inv);
            srow[c] = e;
            sum += e;
        }
        sum += __shfl_xor(sum, 16, 32);
        float rs = 1.0f / sum;
        for (int c = half; c < NN; c += 2)
            Psm[l16 * NN + c] = bf1(srow[c] * rs);
    }

    // ---- pass 2: O = P @ x2[b] ----
    v16bf pa[KT];
#pragma unroll
    for (int kt = 0; kt < KT; ++kt) pa[kt] = load_a_bf16row(&Psm[l16 * NN], kt * 32, half);

    for (int nt = 0; nt < NT; ++nt) {
        v8f acc = {};
        if constexpr (TRANS) {
            const unsigned short* trow = x2t + ((size_t)b * NN + nt * 16 + l16) * NN;
#pragma unroll
            for (int kt = 0; kt < KT; ++kt) {
                v16bf bfrag = load_b_bf16rows(trow, kt * 32, half);
                acc = __builtin_amdgcn_wmma_f32_16x16x32_bf16(false, pa[kt], false, bfrag,
                                                              (short)0, acc, false, false);
            }
        } else {
            const float* bcol = x2b + nt * 16 + l16;
#pragma unroll
            for (int kt = 0; kt < KT; ++kt) {
                v16bf bfrag = load_b_f32cols(bcol, kt * 32, half);
                acc = __builtin_amdgcn_wmma_f32_16x16x32_bf16(false, pa[kt], false, bfrag,
                                                              (short)0, acc, false, false);
            }
        }
        float* orow = out + ((size_t)b * NN + i0) * NN + nt * 16 + l16;
#pragma unroll
        for (int r = 0; r < 8; ++r)
            orow[(size_t)(r + half * 8) * NN] = acc[r];
    }
}

extern "C" void kernel_launch(void* const* d_in, const int* in_sizes, int n_in,
                              void* d_out, int out_size, void* d_ws, size_t ws_size,
                              hipStream_t stream) {
    (void)in_sizes; (void)n_in; (void)out_size;
    const float* x1   = (const float*)d_in[0];
    const float* x2   = (const float*)d_in[1];
    const float* W    = (const float*)d_in[2];
    const float* bias = (const float*)d_in[3];
    float* out = (float*)d_out;

    const size_t x2t_bytes = (size_t)NN * NN * NN * sizeof(unsigned short); // ~113 MB
    const size_t v1_bytes  = (size_t)NN * NN * sizeof(unsigned short);      // ~288 KB
    const bool big = ws_size >= x2t_bytes + v1_bytes;

    unsigned short* x2t = (unsigned short*)d_ws;
    unsigned short* v1w = big ? (unsigned short*)((char*)d_ws + x2t_bytes)
                              : (unsigned short*)d_ws;

    linear_wmma<<<dim3(NT, NT), 32, 0, stream>>>(x1, W, bias, v1w);
    if (big) {
        transpose_bf16<<<dim3(12, 12, NN), 256, 0, stream>>>(x2, x2t);
        fused_attn<true><<<dim3(NT, NN), 32, 0, stream>>>(x2, x2t, v1w, out);
    } else {
        fused_attn<false><<<dim3(NT, NN), 32, 0, stream>>>(x2, nullptr, v1w, out);
    }
}
